// InceptionScatterAttention_37340445671684
// MI455X (gfx1250) — matrix operand order repaired and checked by reading, not compiled, not run on hardware
//
#include <hip/hip_runtime.h>
#include <hip/hip_bf16.h>
#include <math.h>
#include <stdint.h>

typedef __attribute__((ext_vector_type(16))) _Float16 v16h;
typedef __attribute__((ext_vector_type(8)))  float    v8f;
typedef __attribute__((ext_vector_type(4)))  unsigned int u32x4;
typedef __attribute__((ext_vector_type(8)))  int      i32x8;
typedef __attribute__((ext_vector_type(4)))  int      i32x4;

static constexpr int Bn = 2, Ln = 1024, Cn = 512, SCn = 4, Kn = 16, IKn = 17;
static constexpr int CDn  = Cn / SCn;     // 128
static constexpr int NFPn = 7 + Kn;       // 23
static constexpr int FHn  = SCn * 4;      // 16
static constexpr int TOTn = SCn * Kn;     // 64
static constexpr int MID2n = 16, MID1n = 128;
static constexpr size_t BLC = (size_t)Bn * Ln * Cn;   // 1,048,576
static constexpr int ROWS = Bn * Ln;                  // 2048

__device__ __forceinline__ float d_softplus(float x) {
  return x > 20.f ? x : log1pf(expf(x));
}
__device__ __forceinline__ float d_silu(float x) { return x / (1.f + expf(-x)); }
__device__ __forceinline__ float d_sigmoid(float x) { return 1.f / (1.f + expf(-x)); }

__device__ __forceinline__ v8f wmma_f16(v16h a, v16h b, v8f c) {
  return __builtin_amdgcn_wmma_f32_16x16x32_f16(
      false, a, false, b, (short)0, c, false, false);
}

// A-matrix 16x32 f16, contiguous K: lane<16 -> K 0..15, lane>=16 -> K 16..31
__device__ __forceinline__ v16h load_A(const _Float16* base, int ld, int row0, int col0, int lane) {
  int r = row0 + (lane & 15);
  int c = col0 + ((lane >> 4) << 4);
  const _Float16* p = base + (size_t)r * ld + c;
  v16h a;
#pragma unroll
  for (int i = 0; i < 8; ++i) { a[i] = p[i]; a[8 + i] = p[8 + i]; }
  return a;
}

// Pre-swizzled B tile: 32 lanes x 16 halves, one contiguous v16h per lane.
__device__ __forceinline__ v16h load_Bp(const _Float16* base, int tile, int lane) {
  return ((const v16h*)base)[tile * 32 + lane];
}

// ---------------------------------------------------------------- TDM (D#) 2D load
// Descriptor per CDNA5 ISA ch.8: group0 = {flags/count, lds_addr, global_addr, type},
// group1 = {mask/data_size, dims, tile dims, strides}. data_size=1 -> 2-byte elems.
// This toolchain exposes the 6-arg builtin: (u32x4, i32x8, i32x4, i32x4, i32x8, i32 cpol).
__device__ __forceinline__ void tdm_load_2d_f16(unsigned lds_byte_off, const _Float16* gaddr,
                                                unsigned tile_w, unsigned tile_h,
                                                unsigned tensor_w, unsigned tensor_h,
                                                unsigned row_stride_elems) {
#if __has_builtin(__builtin_amdgcn_tensor_load_to_lds)
  unsigned long long ga = (unsigned long long)(uintptr_t)gaddr;
  u32x4 g0;
  g0[0] = 1u;                                        // count=1 (valid user D#)
  g0[1] = lds_byte_off;                              // lds_addr (bytes)
  g0[2] = (unsigned)(ga & 0xffffffffu);              // global_addr[31:0]
  g0[3] = (unsigned)((ga >> 32) & 0x01ffffffu) | (2u << 30);  // addr[56:32] | type=2
  i32x8 g1;
  g1[0] = (int)(1u << 16);                           // workgroup_mask=0, data_size=1 (2B)
  g1[1] = (int)((tensor_w & 0xffffu) << 16);         // tensor_dim0[15:0] at bit48
  g1[2] = (int)(((tensor_w >> 16) & 0xffffu) | ((tensor_h & 0xffffu) << 16));
  g1[3] = (int)(((tensor_h >> 16) & 0xffffu) | ((tile_w & 0xffffu) << 16));
  g1[4] = (int)(tile_h & 0xffffu);                   // tile_dim1; tile_dim2=0
  g1[5] = (int)row_stride_elems;                     // tensor_dim0_stride[31:0]
  g1[6] = 0;
  g1[7] = 0;
  i32x4 z4 = {0, 0, 0, 0};
  i32x8 z8 = {0, 0, 0, 0, 0, 0, 0, 0};
  __builtin_amdgcn_tensor_load_to_lds(g0, g1, z4, z4, z8, 0);
#endif
}

// ---------------------------------------------------------------- zero init
__global__ void kzero(float* p, size_t n) {
  for (size_t i = (size_t)blockIdx.x * blockDim.x + threadIdx.x; i < n;
       i += (size_t)gridDim.x * blockDim.x)
    p[i] = 0.f;
}

// ------------------- weight prep: pre-swizzle all B operands + conv kernel taps
// Packed B layout: tile t, lane l, halves h -> base[(t*32 + l)*16 + h]
// where lane l supplies B[k0 + ((l>>4)<<4) + h][n0 + (l&15)].
__global__ void k_prep(const float* fw1, const float* fw2, const float* fw3, const float* vw,
                       const float* bs, const float* bbf, const float* bbb,
                       _Float16* w1p, _Float16* w2p, _Float16* w3p, _Float16* vwp, float* kern) {
  int i0 = blockIdx.x * blockDim.x + threadIdx.x;
  int stride = gridDim.x * blockDim.x;
  // film_w1: K=512, N=16 -> 16 K-tiles
  for (int t = i0; t < 16 * 512; t += stride) {
    int tile = t >> 9, r = t & 511, lane = r >> 4, h = r & 15;
    int k = tile * 32 + ((lane >> 4) << 4) + h;
    int n = lane & 15;
    w1p[t] = (_Float16)fw1[k * FHn + n];
  }
  // film_w2: 16x16, K padded to 32 -> 1 tile
  for (int t = i0; t < 512; t += stride) {
    int lane = t >> 4, h = t & 15;
    int k = ((lane >> 4) << 4) + h;
    int n = lane & 15;
    w2p[t] = (_Float16)((k < FHn) ? fw2[k * FHn + n] : 0.f);
  }
  // film_w3: 16x92 -> 6 N-tiles, K padded to 32
  for (int t = i0; t < 6 * 512; t += stride) {
    int tile = t >> 9, r = t & 511, lane = r >> 4, h = r & 15;
    int k = ((lane >> 4) << 4) + h;
    int n = tile * 16 + (lane & 15);
    w3p[t] = (_Float16)((k < FHn && n < SCn * NFPn) ? fw3[k * (SCn * NFPn) + n] : 0.f);
  }
  // value_w: per s, B[k=d][n=o] = value_w[s][o][d]; tiles [s][kc(4)][nt(8)]
  for (int t = i0; t < 128 * 512; t += stride) {
    int tile = t >> 9, r = t & 511, lane = r >> 4, h = r & 15;
    int s = tile >> 5, rem = tile & 31, kc = rem >> 3, nt = rem & 7;
    int d = kc * 32 + ((lane >> 4) << 4) + h;
    int o = nt * 16 + (lane & 15);
    vwp[t] = (_Float16)vw[(size_t)s * CDn * CDn + o * CDn + d];
  }
  if (i0 < SCn) {
    int s = i0;
    float st0 = d_softplus(bs[s]);
    float sf = fmaxf(st0 * d_softplus(bbf[s]), 0.5f);
    float sb = fmaxf(st0 * d_softplus(bbb[s]), 0.5f);
    float tmp[IKn]; float sum = 0.f;
#pragma unroll
    for (int j = 0; j < IKn; ++j) {
      float ik = (float)(j - IKn / 2);
      float sig = (ik >= 0.f) ? sf : sb;
      float g = expf(-ik * ik / (2.f * sig * sig));
      tmp[j] = g; sum += g;
    }
    sum = fmaxf(sum, 1e-6f);
#pragma unroll
    for (int j = 0; j < IKn; ++j) kern[s * IKn + j] = tmp[j] / sum;
  }
}

// -------------------------------------- x+pos, depthwise conv, RoPE, f16 copies
__global__ void k_context(const float* x, const float* kern,
                          float* ctxr, _Float16* ctxh, _Float16* xrh) {
  int row = blockIdx.x;             // b*L + l
  int b = row / Ln, l = row % Ln;
  int t = threadIdx.x;              // 0..255 -> channels t and t+256
  int c1 = t, c2 = t + 256;
  float fq1 = powf(10000.f, -(2.f * (float)(c1 >> 1)) / (float)Cn);
  float fq2 = powf(10000.f, -(2.f * (float)(c2 >> 1)) / (float)Cn);
  const float* kr1 = kern + (c1 >> 7) * IKn;
  const float* kr2 = kern + (c2 >> 7) * IKn;
  float acc1 = 0.f, acc2 = 0.f, xl1 = 0.f, xl2 = 0.f;
#pragma unroll
  for (int j = 0; j < IKn; ++j) {
    int lp = l + j - IKn / 2;
    if (lp < 0 || lp >= Ln) continue;
    float a1 = fq1 * (float)lp, a2 = fq2 * (float)lp;
    float p1 = (c1 & 1) ? cosf(a1) : sinf(a1);
    float p2 = (c2 & 1) ? cosf(a2) : sinf(a2);
    float xp1 = x[((size_t)b * Ln + lp) * Cn + c1] + p1;
    float xp2 = x[((size_t)b * Ln + lp) * Cn + c2] + p2;
    acc1 += kr1[j] * xp1; acc2 += kr2[j] * xp2;
    if (j == IKn / 2) { xl1 = xp1; xl2 = xp2; }
  }
  float ar = (float)l * powf(10000.f, -(float)t / 256.f);
  float cr = cosf(ar), sr = sinf(ar);
  size_t base = (size_t)row * Cn;
  float o1 = acc1 * cr - acc2 * sr, o2 = acc1 * sr + acc2 * cr;
  ctxr[base + c1] = o1;            ctxr[base + c2] = o2;
  ctxh[base + c1] = (_Float16)o1;  ctxh[base + c2] = (_Float16)o2;
  float r1 = xl1 * cr - xl2 * sr,  r2 = xl1 * sr + xl2 * cr;
  xrh[base + c1] = (_Float16)r1;   xrh[base + c2] = (_Float16)r2;
}

// ---------------- FiLM 3-layer MLP fused: TDM-staged A panel + WMMA + LDS chain
__global__ void k_film(const _Float16* ctxh, const _Float16* w1p, const _Float16* w2p,
                       const _Float16* w3p, const float* b1, const float* b2,
                       const float* b3, float* film) {
  __shared__ _Float16 Abuf[128][32];   // first LDS object -> byte offset 0 for TDM
  __shared__ _Float16 H1[128][32];
  __shared__ _Float16 H2[128][32];
  int tid = threadIdx.x, lane = tid & 31, w = tid >> 5;   // 8 waves
  int m0 = blockIdx.x * 128;
  for (int t = tid; t < 128 * 32; t += 256) {
    Abuf[t >> 5][t & 31] = (_Float16)0.f;   // IR-visible writer for Abuf
    H1[t >> 5][t & 31] = (_Float16)0.f;
    H2[t >> 5][t & 31] = (_Float16)0.f;
  }
  __syncthreads();
  int col0 = lane & 15, rbase = (lane >> 4) << 3;
  // Layer 1: wave w -> M-tile w (16 rows), N-tile 0. K = 512 in 16 chunks of 32,
  // each chunk DMA'd into LDS by the Tensor Data Mover.
  {
    v8f acc = {};
    for (int kc = 0; kc < 16; ++kc) {
#if __has_builtin(__builtin_amdgcn_tensor_load_to_lds)
      if (w == 0) {
        tdm_load_2d_f16(/*lds_off=*/0u,
                        ctxh + (size_t)m0 * Cn + kc * 32,
                        /*tile_w=*/32, /*tile_h=*/128,
                        /*tensor_w=*/Cn, /*tensor_h=*/ROWS,
                        /*row_stride=*/Cn);
        __builtin_amdgcn_s_wait_tensorcnt(0);
      }
#else
      for (int t = tid; t < 128 * 32; t += 256) {
        int rr = t >> 5, cc = t & 31;
        Abuf[rr][cc] = ctxh[(size_t)(m0 + rr) * Cn + kc * 32 + cc];
      }
#endif
      __syncthreads();
      // TDM writes LDS behind the compiler's back: force re-materialization of
      // the Abuf loads each iteration.
      asm volatile("" ::: "memory");
      v16h a = load_A(&Abuf[0][0], 32, w * 16, 0, lane);
      v16h b = load_Bp(w1p, kc, lane);
      acc = wmma_f16(a, b, acc);
      __syncthreads();
    }
#pragma unroll
    for (int r = 0; r < 8; ++r) {
      float h = acc[r] + b1[col0];
      H1[w * 16 + rbase + r][col0] = (_Float16)d_silu(h);
    }
  }
  __syncthreads();
  // Layer 2: (128x16)@(16x16), K padded to 32 with zeros.
  {
    v8f acc = {};
    v16h a = load_A(&H1[0][0], 32, w * 16, 0, lane);
    v16h b = load_Bp(w2p, 0, lane);
    acc = wmma_f16(a, b, acc);
#pragma unroll
    for (int r = 0; r < 8; ++r) {
      float h = acc[r] + b2[col0];
      H2[w * 16 + rbase + r][col0] = (_Float16)d_silu(h);
    }
  }
  __syncthreads();
  // Layer 3: (128x16)@(16x96[92 valid]); 8 Mtiles x 6 Ntiles = 48 tasks.
  for (int task = w; task < 48; task += 8) {
    int mt = task / 6, nt = task % 6;
    v8f acc = {};
    v16h a = load_A(&H2[0][0], 32, mt * 16, 0, lane);
    v16h b = load_Bp(w3p, nt, lane);
    acc = wmma_f16(a, b, acc);
    int col = nt * 16 + col0;
    if (col < SCn * NFPn) {
#pragma unroll
      for (int r = 0; r < 8; ++r)
        film[(size_t)(m0 + mt * 16 + rbase + r) * 96 + col] = acc[r] + b3[col];
    }
  }
}

// ---------------------------------------------- per-(b,l,s) params + tanh bias
__global__ void k_params(const float* film, const float* bd, const float* bs,
                         const float* bbf, const float* bbb, const float* bstr,
                         const float* baf, const float* bab, const float* samp,
                         float* params, float* sb_pre, float* pooled) {
  int gid = blockIdx.x * blockDim.x + threadIdx.x;
  if (gid >= Bn * Ln * SCn) return;
  int s = gid % SCn, row = gid / SCn, b = row / Ln;
  const float* f = film + (size_t)row * 96 + s * NFPn;
  float* pp = params + (size_t)gid * 8;
  float d = bd[s] + f[0];
  pp[0] = fminf(fmaxf(d, -32.f), 32.f);
  pp[1] = d_softplus(bs[s] + f[1]);
  pp[2] = d_softplus(bbf[s] + f[2]);
  pp[3] = d_softplus(bbb[s] + f[3]);
  pp[4] = d_softplus(bstr[s] + f[4]);
  pp[5] = d_softplus(baf[s] + f[5]);
  pp[6] = d_softplus(bab[s] + f[6]);
  pp[7] = 0.f;
#pragma unroll
  for (int k = 0; k < Kn; ++k) {
    float v = tanhf(samp[s * Kn + k] + f[7 + k]);
    sb_pre[(size_t)row * TOTn + s * Kn + k] = v;
    atomicAdd(&pooled[b * TOTn + s * Kn + k], v * (1.f / (float)Ln));
  }
}

// -------------------------------------------------------- global SE gate (sg)
__global__ void k_sg(const float* pooled, const float* w1g, const float* b1g,
                     const float* w2g, const float* b2g, float* sg) {
  __shared__ float mid[MID2n];
  int b = blockIdx.x, t = threadIdx.x;
  if (t < MID2n) {
    float a = b1g[t];
    for (int j = 0; j < TOTn; ++j) a += pooled[b * TOTn + j] * w1g[j * MID2n + t];
    mid[t] = d_silu(a);
  }
  __syncthreads();
  float a = b2g[t];
  for (int m = 0; m < MID2n; ++m) a += mid[m] * w2g[m * TOTn + t];
  sg[b * TOTn + t] = d_sigmoid(a);
}

// ----------------------------------------- local SE gate (sl) + final sb
__global__ void k_slsb(const float* sb_pre, const float* sg, const float* w1l,
                       const float* b1l, const float* w2l, const float* b2l, float* sb) {
  __shared__ float sh[TOTn];
  __shared__ float mid[MID2n];
  int row = blockIdx.x, b = row / Ln, t = threadIdx.x;
  float g = sb_pre[(size_t)row * TOTn + t] * sg[b * TOTn + t];
  sh[t] = g;
  __syncthreads();
  if (t < MID2n) {
    float a = b1l[t];
    for (int j = 0; j < TOTn; ++j) a += sh[j] * w1l[j * MID2n + t];
    mid[t] = d_silu(a);
  }
  __syncthreads();
  float a = b2l[t];
  for (int m = 0; m < MID2n; ++m) a += mid[m] * w2l[m * TOTn + t];
  sb[(size_t)row * TOTn + t] = g * d_sigmoid(a);
}

// ------------------------------- vals: 4 x (2048x128)@(128x128) via WMMA
__global__ void k_vals(const _Float16* xrh, const _Float16* vwp, float* vals) {
  int tid = threadIdx.x, lane = tid & 31, w = tid >> 5;
  int s = blockIdx.y;
  int m0 = blockIdx.x * 32;
  int col0 = lane & 15, rbase = (lane >> 4) << 3;
  for (int task = w; task < 16; task += 8) {        // 2 Mtiles x 8 Ntiles
    int mt = task >> 3, nt = task & 7;
    v8f acc = {};
#pragma unroll
    for (int kc = 0; kc < 4; ++kc) {
      if (kc + 1 < 4)
        __builtin_prefetch(xrh + (size_t)(m0 + mt * 16 + (lane & 15)) * Cn + s * CDn + (kc + 1) * 32, 0, 3);
      v16h a = load_A(xrh + (size_t)m0 * Cn + s * CDn, Cn, mt * 16, kc * 32, lane);
      v16h b = load_Bp(vwp, s * 32 + kc * 8 + nt, lane);
      acc = wmma_f16(a, b, acc);
    }
    int col = nt * 16 + col0;
#pragma unroll
    for (int r = 0; r < 8; ++r)
      vals[(size_t)(m0 + mt * 16 + rbase + r) * Cn + s * CDn + col] = acc[r];
  }
}

// ------------- bilinear gather + value_bias dot + env + bilinear scatter-add
__global__ void k_scatter(const float* ctxr, const float* params, const float* sb,
                          const float* vals, float* num, float* den) {
  __shared__ float ctxl[Cn];
  __shared__ float vbS[TOTn], envS[TOTn], fracS[TOTn];
  __shared__ int i0S[TOTn], i1S[TOTn];
  int row = blockIdx.x, b = row / Ln, l = row % Ln;
  int tid = threadIdx.x;
  for (int c = tid; c < Cn; c += 256) ctxl[c] = ctxr[(size_t)row * Cn + c];
  __syncthreads();

  int p = tid >> 2, q = tid & 3;           // 64 (s,k) pairs, 4 lanes each
  int s = p >> 4, kk = p & 15;
  const float* pr = params + ((size_t)row * SCn + s) * 8;
  float deform = pr[0], strd = pr[1], bf = pr[2], bb = pr[3];
  float stren = pr[4], af = pr[5], ab = pr[6];
  int kv = kk - Kn / 2;
  float ka = fabsf((float)kv);
  float kas = ka > 0.f ? ka : 1.f;
  float pw = ka > 0.f ? powf(kas, (kv >= 0) ? bf : bb) : 0.f;
  float warp = ((kv >= 0) ? pw : -pw) * strd;
  float pos = fminf(fmaxf((float)l + deform + warp, 0.f), (float)(Ln - 1));
  float p0 = floorf(pos);
  float frac = pos - p0;
  int i0 = (int)p0; i0 = i0 < 0 ? 0 : (i0 > Ln - 1 ? Ln - 1 : i0);
  int i1 = i0 + 1 > Ln - 1 ? Ln - 1 : i0 + 1;
  float ea = ka > 0.f ? powf(kas, (kv >= 0) ? af : ab) : 0.f;
  float env = stren / (1.f + ea) * (1.f + sb[(size_t)row * TOTn + p]);

  const float* g0 = ctxr + ((size_t)b * Ln + i0) * Cn;
  const float* g1 = ctxr + ((size_t)b * Ln + i1) * Cn;
  float sum = 0.f;
  for (int c = q * 128; c < q * 128 + 128; ++c) {
    float a0 = g0[c], a1 = g1[c];
    sum += (a0 + frac * (a1 - a0)) * ctxl[c];
  }
  sum += __shfl_down(sum, 1);
  sum += __shfl_down(sum, 2);
  if (q == 0) { vbS[p] = sum; envS[p] = env; fracS[p] = frac; i0S[p] = i0; i1S[p] = i1; }
  __syncthreads();

  const float* vrow = vals + (size_t)row * Cn;
  for (int idx = tid; idx < TOTn * CDn; idx += 256) {
    int pp = idx >> 7, o = idx & 127;
    int ss = pp >> 4;
    float e = envS[pp], fr = fracS[pp];
    float v = (vrow[ss * CDn + o] + vbS[pp]) * e;
    atomicAdd(&num[((size_t)b * Ln + i0S[pp]) * Cn + ss * CDn + o], (1.f - fr) * v);
    atomicAdd(&num[((size_t)b * Ln + i1S[pp]) * Cn + ss * CDn + o], fr * v);
  }
  if (tid < TOTn) {
    int pp = tid, ss = pp >> 4;
    atomicAdd(&den[((size_t)b * Ln + i0S[pp]) * SCn + ss], (1.f - fracS[pp]) * envS[pp]);
    atomicAdd(&den[((size_t)b * Ln + i1S[pp]) * SCn + ss], fracS[pp] * envS[pp]);
  }
}

// --------------------------------------------- pooled output (mean over L)
__global__ void k_po(const float* num, const float* den, float* po) {
  int idx = blockIdx.x * blockDim.x + threadIdx.x;
  if (idx >= Bn * Cn) return;
  int b = idx / Cn, c = idx % Cn, s = c >> 7;
  float sum = 0.f;
  for (int l = 0; l < Ln; ++l) {
    size_t r = (size_t)b * Ln + l;
    sum += num[r * Cn + c] / fmaxf(den[r * SCn + s], 1e-6f);
  }
  po[idx] = sum / (float)Ln;
}

// ------------------------------------------------------------- SE1 gate
__global__ void k_se1(const float* po, const float* w1, const float* b1,
                      const float* w2, const float* b2, float* s1g) {
  __shared__ float mid[MID1n];
  int b = blockIdx.x, t = threadIdx.x;  // 128 threads
  float a = b1[t];
  for (int c = 0; c < Cn; ++c) a += po[b * Cn + c] * w1[c * MID1n + t];
  mid[t] = d_silu(a);
  __syncthreads();
  for (int c = t; c < Cn; c += MID1n) {
    float o = b2[c];
    for (int m = 0; m < MID1n; ++m) o += mid[m] * w2[m * Cn + c];
    s1g[b * Cn + c] = d_sigmoid(o);
  }
}

// ----------------------------------------------- normalize + gate + residual
__global__ void k_final(const float* num, const float* den, const float* s1g,
                        const float* x, float* out) {
  size_t total = BLC;
  for (size_t idx = (size_t)blockIdx.x * blockDim.x + threadIdx.x; idx < total;
       idx += (size_t)gridDim.x * blockDim.x) {
    size_t r = idx / Cn; int c = (int)(idx % Cn); int s = c >> 7; int b = (int)(r / Ln);
    float v = num[idx] / fmaxf(den[r * SCn + s], 1e-6f);
    out[idx] = v * s1g[b * Cn + c] + x[idx];
  }
}

extern "C" void kernel_launch(void* const* d_in, const int* in_sizes, int n_in,
                              void* d_out, int out_size, void* d_ws, size_t ws_size,
                              hipStream_t stream) {
  const float* x    = (const float*)d_in[0];
  const float* bd   = (const float*)d_in[1];
  const float* bs   = (const float*)d_in[2];
  const float* bbf  = (const float*)d_in[3];
  const float* bbb  = (const float*)d_in[4];
  const float* bstr = (const float*)d_in[5];
  const float* baf  = (const float*)d_in[6];
  const float* bab  = (const float*)d_in[7];
  const float* samp = (const float*)d_in[8];
  const float* fw1  = (const float*)d_in[9];
  const float* fb1  = (const float*)d_in[10];
  const float* fw2  = (const float*)d_in[11];
  const float* fb2  = (const float*)d_in[12];
  const float* fw3  = (const float*)d_in[13];
  const float* fb3  = (const float*)d_in[14];
  const float* vw   = (const float*)d_in[15];
  const float* w1g  = (const float*)d_in[16];
  const float* b1g  = (const float*)d_in[17];
  const float* w2g  = (const float*)d_in[18];
  const float* b2g  = (const float*)d_in[19];
  const float* w1l  = (const float*)d_in[20];
  const float* b1l  = (const float*)d_in[21];
  const float* w2l  = (const float*)d_in[22];
  const float* b2l  = (const float*)d_in[23];
  const float* s1w1 = (const float*)d_in[24];
  const float* s1b1 = (const float*)d_in[25];
  const float* s1w2 = (const float*)d_in[26];
  const float* s1b2 = (const float*)d_in[27];

  char* wsp = (char*)d_ws;
  size_t off = 0;
  auto alloc = [&](size_t bytes) -> char* {
    char* p = wsp + off;
    off = (off + bytes + 255) & ~(size_t)255;
    return p;
  };
  float*     ctxr   = (float*)alloc(BLC * 4);
  _Float16*  ctxh   = (_Float16*)alloc(BLC * 2);
  _Float16*  xrh    = (_Float16*)alloc(BLC * 2);
  float*     film   = (float*)alloc((size_t)ROWS * 96 * 4);
  _Float16*  w1p    = (_Float16*)alloc((size_t)16 * 512 * 2);
  _Float16*  w2p    = (_Float16*)alloc((size_t)512 * 2);
  _Float16*  w3p    = (_Float16*)alloc((size_t)6 * 512 * 2);
  _Float16*  vwp    = (_Float16*)alloc((size_t)128 * 512 * 2);
  float*     kern   = (float*)alloc((size_t)SCn * IKn * 4);
  float*     params = (float*)alloc((size_t)ROWS * SCn * 8 * 4);
  float*     sb_pre = (float*)alloc((size_t)ROWS * TOTn * 4);
  float*     sbv    = (float*)alloc((size_t)ROWS * TOTn * 4);
  float*     pooled = (float*)alloc((size_t)Bn * TOTn * 4);
  float*     sg     = (float*)alloc((size_t)Bn * TOTn * 4);
  float*     vals   = (float*)alloc(BLC * 4);
  float*     num    = (float*)alloc(BLC * 4);
  float*     den    = (float*)alloc((size_t)ROWS * SCn * 4);
  float*     po     = (float*)alloc((size_t)Bn * Cn * 4);
  float*     s1g    = (float*)alloc((size_t)Bn * Cn * 4);

  kzero<<<256, 256, 0, stream>>>(num, BLC);
  kzero<<<32, 256, 0, stream>>>(den, (size_t)ROWS * SCn);
  kzero<<<1, 128, 0, stream>>>(pooled, (size_t)Bn * TOTn);

  k_prep<<<128, 256, 0, stream>>>(fw1, fw2, fw3, vw, bs, bbf, bbb, w1p, w2p, w3p, vwp, kern);
  k_context<<<ROWS, 256, 0, stream>>>(x, kern, ctxr, ctxh, xrh);
  k_film<<<ROWS / 128, 256, 0, stream>>>(ctxh, w1p, w2p, w3p, fb1, fb2, fb3, film);
  k_params<<<(Bn * Ln * SCn + 255) / 256, 256, 0, stream>>>(
      film, bd, bs, bbf, bbb, bstr, baf, bab, samp, params, sb_pre, pooled);
  k_sg<<<Bn, TOTn, 0, stream>>>(pooled, w1g, b1g, w2g, b2g, sg);
  k_slsb<<<ROWS, TOTn, 0, stream>>>(sb_pre, sg, w1l, b1l, w2l, b2l, sbv);
  k_vals<<<dim3(ROWS / 32, SCn), 256, 0, stream>>>(xrh, vwp, vals);
  k_scatter<<<ROWS, 256, 0, stream>>>(ctxr, params, sbv, vals, num, den);
  k_po<<<(Bn * Cn + 255) / 256, 256, 0, stream>>>(num, den, po);
  k_se1<<<Bn, MID1n, 0, stream>>>(po, s1w1, s1b1, s1w2, s1b2, s1g);
  k_final<<<ROWS, 256, 0, stream>>>(num, den, s1g, x, (float*)d_out);
}
